// GatedDeltaNetBlock_71201967833670
// MI455X (gfx1250) — matrix operand order repaired and checked by reading, not compiled, not run on hardware
//
#include <hip/hip_runtime.h>

typedef unsigned short u16;
typedef __attribute__((ext_vector_type(16))) __bf16 v16bf;
typedef __attribute__((ext_vector_type(8)))  float  v8f;

// ---------------- helpers ----------------
__device__ __forceinline__ float sigmoidf_(float x) { return 1.f / (1.f + __expf(-x)); }
__device__ __forceinline__ float siluf_(float x)    { return x * sigmoidf_(x); }

__device__ __forceinline__ u16 f32_to_bf16_bits(float f) {
  union { float f; unsigned u; } a; a.f = f;
  unsigned u = a.u;
  unsigned r = u + 0x7fffu + ((u >> 16) & 1u);   // RNE
  return (u16)(r >> 16);
}
__device__ __forceinline__ __bf16 bits_to_bf16(u16 s) {
  union { u16 s; __bf16 b; } x; x.s = s; return x.b;
}

// Load 16 contiguous bf16 (32 bytes, 2x b128) into a v16bf fragment.
__device__ __forceinline__ v16bf load_frag16(const u16* __restrict__ p) {
  uint4 p0 = *reinterpret_cast<const uint4*>(p);
  uint4 p1 = *reinterpret_cast<const uint4*>(p + 8);
  unsigned pw[8] = {p0.x, p0.y, p0.z, p0.w, p1.x, p1.y, p1.z, p1.w};
  v16bf f;
#pragma unroll
  for (int i = 0; i < 8; ++i) {
    f[2 * i]     = bits_to_bf16((u16)(pw[i] & 0xffffu));
    f[2 * i + 1] = bits_to_bf16((u16)(pw[i] >> 16));
  }
  return f;
}

// ---------------- f32 -> bf16 conversion (flat, for A-side activations) ----------------
__global__ void cvt_bf16_kernel(const float* __restrict__ in, u16* __restrict__ out, int n) {
  int i = blockIdx.x * blockDim.x + threadIdx.x;
  if (i < n) out[i] = f32_to_bf16_bits(in[i]);
}

// ---------------- f32 -> bf16 B-operand pack into WMMA fragment order ----------------
// out[(k0/32)*N*32 + col*32 + slot]:
//   slot 0..15  -> K = k0 + {0..15}   (consumed by lanes 0..15)
//   slot 16..31 -> K = k0 + {16..31}  (consumed by lanes 16..31)
__global__ void pack_b_kernel(const float* __restrict__ B, u16* __restrict__ out,
                              int K, int N) {
  int idx = blockIdx.x * blockDim.x + threadIdx.x;
  if (idx >= K * N) return;
  const int kblk = idx / (N * 32);
  const int rem  = idx - kblk * (N * 32);
  const int col  = rem >> 5;
  const int slot = rem & 31;
  const int s2   = slot & 15;
  const int kb   = (slot >= 16) ? 8 : 0;
  const int km   = slot;
  out[idx] = f32_to_bf16_bits(B[(size_t)(kblk * 32 + km) * N + col]);
}

// ---------------- zero-centered RMSNorm (fp32 out + bf16 out) ----------------
__global__ __launch_bounds__(256)
void zrms_kernel(const float* __restrict__ in, const float* __restrict__ w,
                 float* __restrict__ outf, u16* __restrict__ outb) {
  __shared__ float red[256];
  const int s = blockIdx.x, t = threadIdx.x;
  const float* row = in + s * 1024;
  float ss = 0.f;
  for (int j = t; j < 1024; j += 256) { float v = row[j]; ss += v * v; }
  red[t] = ss; __syncthreads();
  for (int o = 128; o > 0; o >>= 1) { if (t < o) red[t] += red[t + o]; __syncthreads(); }
  const float r = rsqrtf(red[0] / 1024.f + 1e-6f);
  for (int j = t; j < 1024; j += 256) {
    float v = row[j] * r * (1.f + w[j]);
    outf[s * 1024 + j] = v;
    outb[s * 1024 + j] = f32_to_bf16_bits(v);
  }
}

// ---------------- bf16 WMMA GEMM: C[M,N] = A[M,K]*Bp[K,N] (+addsrc) ----------------
// A row-major bf16; Bp packed by pack_b_kernel. block = 256 threads (8 waves);
// block tile 256x64; wave tile 32x64 (2 A-frags x 4 B-tiles = 8 wmma / K-step).
// All fragment loads are 2x b128; B fragments reused across both A fragments.
template <bool ADD>
__global__ __launch_bounds__(256)
void gemm_bf16_wmma(const u16* __restrict__ A, const u16* __restrict__ Bp,
                    const float* __restrict__ addsrc, float* __restrict__ C,
                    int M, int N, int K) {
  const int lane = threadIdx.x & 31;
  const int wv   = threadIdx.x >> 5;
  const int bm   = blockIdx.y * 256;
  const int bn   = blockIdx.x * 64;
  const int rowA0 = bm + wv * 32 + (lane & 15);
  const int kb    = (lane >= 16) ? 16 : 0;   // A-side K offset
  const int khalf = (lane >= 16) ? 16 : 0;   // B-side packed slot offset
  const int colb  = bn + (lane & 15);

  v8f acc[2][4] = {};
  for (int k0 = 0; k0 < K; k0 += 32) {
    v16bf a0 = load_frag16(A + (size_t)rowA0 * K + k0 + kb);
    v16bf a1 = load_frag16(A + (size_t)(rowA0 + 16) * K + k0 + kb);
    const u16* bbase = Bp + ((size_t)(k0 >> 5) * N + colb) * 32 + khalf;
#pragma unroll
    for (int t = 0; t < 4; ++t) {
      v16bf b = load_frag16(bbase + t * 512);
      acc[0][t] = __builtin_amdgcn_wmma_f32_16x16x32_bf16(
          false, a0, false, b, (short)0, acc[0][t], false, false);
      acc[1][t] = __builtin_amdgcn_wmma_f32_16x16x32_bf16(
          false, a1, false, b, (short)0, acc[1][t], false, false);
    }
  }
  const int rowc0 = bm + wv * 32 + ((lane >= 16) ? 8 : 0);
#pragma unroll
  for (int r = 0; r < 2; ++r) {
#pragma unroll
    for (int t = 0; t < 4; ++t) {
      const int col = colb + t * 16;
#pragma unroll
      for (int j = 0; j < 8; ++j) {
        const int row = rowc0 + r * 16 + j;
        float v = acc[r][t][j];
        if (ADD) v += addsrc[(size_t)row * N + col];
        C[(size_t)row * N + col] = v;
      }
    }
  }
}

// ---------------- small GEMM x @ W_ba -> [2048,16] ----------------
__global__ void ba_kernel(const float* __restrict__ x, const float* __restrict__ Wba,
                          float* __restrict__ ba) {
  int idx = blockIdx.x * blockDim.x + threadIdx.x;
  if (idx >= 2048 * 16) return;
  int s = idx >> 4, e = idx & 15;
  float acc = 0.f;
  for (int d = 0; d < 1024; ++d) acc += x[s * 1024 + d] * Wba[d * 16 + e];
  ba[idx] = acc;
}

// ---------------- causal depthwise conv (K=4) + SiLU over first 2048 chans ----------------
__global__ void conv_silu_kernel(const float* __restrict__ qkvz, const float* __restrict__ cw,
                                 float* __restrict__ mixed) {
  int idx = blockIdx.x * blockDim.x + threadIdx.x;
  if (idx >= 2048 * 2048) return;
  int s = idx >> 11, c = idx & 2047;
  float acc = 0.f;
#pragma unroll
  for (int j = 0; j < 4; ++j) {
    int sp = s - 3 + j;
    if (sp >= 0) acc += qkvz[(size_t)sp * 3072 + c] * cw[c * 4 + j];
  }
  mixed[idx] = siluf_(acc);
}

// ---------------- per-(s,head): l2-norm q,k (with repeat 4->8), beta, g ----------------
__global__ void prep_kernel(const float* __restrict__ mixed, const float* __restrict__ ba,
                            const float* __restrict__ A_log, const float* __restrict__ dtb,
                            float* __restrict__ qn, float* __restrict__ kn,
                            float* __restrict__ gv, float* __restrict__ bv) {
  int idx = blockIdx.x * blockDim.x + threadIdx.x;
  if (idx >= 2048 * 8) return;
  int s = idx >> 3, h = idx & 7, hk = h >> 1;
  const float* qp = mixed + (size_t)s * 2048 + hk * 128;
  const float* kp = mixed + (size_t)s * 2048 + 512 + hk * 128;
  float sq = 0.f, sk = 0.f;
  for (int d = 0; d < 128; ++d) { sq += qp[d] * qp[d]; sk += kp[d] * kp[d]; }
  const float rq = rsqrtf(sq + 1e-6f) * 0.08838834764831845f;  // * DK^-0.5
  const float rk = rsqrtf(sk + 1e-6f);
  float* qo = qn + (size_t)idx * 128;
  float* ko = kn + (size_t)idx * 128;
  for (int d = 0; d < 128; ++d) { qo[d] = qp[d] * rq; ko[d] = kp[d] * rk; }
  const float b = ba[s * 16 + h], a = ba[s * 16 + 8 + h];
  bv[idx] = sigmoidf_(b);
  const float x = a + dtb[h];
  const float sp = (x > 20.f) ? x : log1pf(__expf(x));
  gv[idx] = -__expf(A_log[h]) * sp;
}

// ---------------- gated delta rule: state column per thread, all in VGPRs ----------------
__global__ __launch_bounds__(128)
void delta_kernel(const float* __restrict__ qn, const float* __restrict__ kn,
                  const float* __restrict__ mixed, const float* __restrict__ gv,
                  const float* __restrict__ bv, float* __restrict__ core) {
  const int h = blockIdx.x;    // head 0..7
  const int v = threadIdx.x;   // state column 0..127
  float S[128];
#pragma unroll
  for (int k = 0; k < 128; ++k) S[k] = 0.f;
#pragma unroll 1
  for (int s = 0; s < 2048; ++s) {
    const int base = (s * 8 + h) * 128;
    const float dec  = __expf(gv[s * 8 + h]);
    const float beta = bv[s * 8 + h];
    const float vt   = mixed[(size_t)s * 2048 + 1024 + h * 128 + v];
    float kv = 0.f;
#pragma unroll
    for (int k = 0; k < 128; ++k) { S[k] *= dec; kv += kn[base + k] * S[k]; }
    const float delta = (vt - kv) * beta;
    float o = 0.f;
#pragma unroll
    for (int k = 0; k < 128; ++k) { S[k] += kn[base + k] * delta; o += qn[base + k] * S[k]; }
    core[base + v] = o;
  }
}

// ---------------- gated RMSNorm per (s,head) + silu(z) gate -> bf16 ----------------
__global__ __launch_bounds__(128)
void gatenorm_kernel(const float* __restrict__ core, const float* __restrict__ qkvz,
                     const float* __restrict__ gnw, u16* __restrict__ outb) {
  __shared__ float red[128];
  const int sh = blockIdx.x;    // s*8+h
  const int d  = threadIdx.x;
  const int s = sh >> 3, h = sh & 7;
  const float c = core[(size_t)sh * 128 + d];
  red[d] = c * c; __syncthreads();
  for (int o = 64; o > 0; o >>= 1) { if (d < o) red[d] += red[d + o]; __syncthreads(); }
  const float r = rsqrtf(red[0] / 128.f + 1e-6f);
  const float z = qkvz[(size_t)s * 3072 + 2048 + h * 128 + d];
  outb[(size_t)s * 1024 + h * 128 + d] = f32_to_bf16_bits(c * r * gnw[d] * siluf_(z));
}

// ---------------- router: softmax + top-2 (first-index tie-break) ----------------
__global__ void router_kernel(const float* __restrict__ x2, const float* __restrict__ Wr,
                              float* __restrict__ route, int* __restrict__ topi) {
  int s = blockIdx.x * blockDim.x + threadIdx.x;
  if (s >= 2048) return;
  float lg[8];
#pragma unroll
  for (int e = 0; e < 8; ++e) lg[e] = 0.f;
  for (int d = 0; d < 1024; ++d) {
    const float xv = x2[(size_t)s * 1024 + d];
#pragma unroll
    for (int e = 0; e < 8; ++e) lg[e] += xv * Wr[d * 8 + e];
  }
  float mx = lg[0];
#pragma unroll
  for (int e = 1; e < 8; ++e) mx = fmaxf(mx, lg[e]);
  float p[8], sum = 0.f;
#pragma unroll
  for (int e = 0; e < 8; ++e) { p[e] = __expf(lg[e] - mx); sum += p[e]; }
#pragma unroll
  for (int e = 0; e < 8; ++e) p[e] /= sum;
  int i0 = 0; float v0 = p[0];
#pragma unroll
  for (int e = 1; e < 8; ++e) if (p[e] > v0) { v0 = p[e]; i0 = e; }
  int i1 = -1; float v1 = -1.f;
#pragma unroll
  for (int e = 0; e < 8; ++e) if (e != i0 && p[e] > v1) { v1 = p[e]; i1 = e; }
  const float wsum = v0 + v1;
  const float w0 = v0 / wsum, w1 = v1 / wsum;
#pragma unroll
  for (int e = 0; e < 8; ++e) route[s * 8 + e] = (e == i0) ? w0 : ((e == i1) ? w1 : 0.f);
  topi[s * 2 + 0] = i0;
  topi[s * 2 + 1] = i1;
}

// ---------------- misc elementwise ----------------
__global__ void silu_mul_kernel(const float* __restrict__ g, const float* __restrict__ u,
                                u16* __restrict__ out, int n) {
  int i = blockIdx.x * blockDim.x + threadIdx.x;
  if (i < n) out[i] = f32_to_bf16_bits(siluf_(g[i]) * u[i]);
}
__global__ void copy_kernel(const float* __restrict__ in, float* __restrict__ out, int n) {
  int i = blockIdx.x * blockDim.x + threadIdx.x;
  if (i < n) out[i] = in[i];
}
__global__ void accum_kernel(float* __restrict__ out, const float* __restrict__ expb,
                             const float* __restrict__ route, int e) {
  int i = blockIdx.x * blockDim.x + threadIdx.x;
  if (i < 2048 * 1024) { int s = i >> 10; out[i] += route[s * 8 + e] * expb[i]; }
}

// ---------------- host launcher ----------------
extern "C" void kernel_launch(void* const* d_in, const int* in_sizes, int n_in,
                              void* d_out, int out_size, void* d_ws, size_t ws_size,
                              hipStream_t stream) {
  (void)in_sizes; (void)n_in; (void)out_size; (void)ws_size;
  const float* hs    = (const float*)d_in[0];
  const float* wln1  = (const float*)d_in[1];
  const float* wln2  = (const float*)d_in[2];
  const float* Wqkvz = (const float*)d_in[3];
  const float* Wba   = (const float*)d_in[4];
  const float* convw = (const float*)d_in[5];
  const float* dtb   = (const float*)d_in[6];
  const float* Alog  = (const float*)d_in[7];
  const float* gnw   = (const float*)d_in[8];
  const float* Wout  = (const float*)d_in[9];
  const float* Wr    = (const float*)d_in[10];
  const float* Wg    = (const float*)d_in[11];
  const float* Wu    = (const float*)d_in[12];
  const float* Wd    = (const float*)d_in[13];

  char* ws = (char*)d_ws;
  size_t off = 0;
  auto alloc = [&](size_t bytes) -> char* {
    char* p = ws + off; off += (bytes + 255) & ~(size_t)255; return p;
  };
  float* x1f   = (float*)alloc((size_t)2048 * 1024 * 4);   // reused as x2
  u16*   x1b   = (u16*)  alloc((size_t)2048 * 1024 * 2);   // reused as x2 bf16
  float* qkvz  = (float*)alloc((size_t)2048 * 3072 * 4);
  float* bab   = (float*)alloc((size_t)2048 * 16 * 4);
  float* mixed = (float*)alloc((size_t)2048 * 2048 * 4);
  float* qn    = (float*)alloc((size_t)2048 * 8 * 128 * 4);
  float* kn    = (float*)alloc((size_t)2048 * 8 * 128 * 4);
  float* gv    = (float*)alloc((size_t)2048 * 8 * 4);
  float* bv    = (float*)alloc((size_t)2048 * 8 * 4);
  float* coreb = (float*)alloc((size_t)2048 * 1024 * 4);
  u16*   coreB = (u16*)  alloc((size_t)2048 * 1024 * 2);
  float* hbuf  = (float*)alloc((size_t)2048 * 1024 * 4);
  u16*   wstg  = (u16*)  alloc((size_t)1024 * 3072 * 2);   // packed bf16 weight staging (max)
  float* gbuf  = (float*)alloc((size_t)2048 * 512 * 4);
  float* ubuf  = (float*)alloc((size_t)2048 * 512 * 4);
  u16*   tmpb  = (u16*)  alloc((size_t)2048 * 512 * 2);
  float* expb  = (float*)alloc((size_t)2048 * 1024 * 4);
  float* route = (float*)alloc((size_t)2048 * 8 * 4);

  float* outf = (float*)d_out;
  int*   topi = (int*)((float*)d_out + (size_t)2048 * 1024);

  const int T = 256;
  auto cdiv = [](int a, int b) { return (a + b - 1) / b; };

  // ---- deltanet sub-block ----
  zrms_kernel<<<2048, 256, 0, stream>>>(hs, wln1, x1f, x1b);
  pack_b_kernel<<<cdiv(1024 * 3072, T), T, 0, stream>>>(Wqkvz, wstg, 1024, 3072);
  gemm_bf16_wmma<false><<<dim3(3072 / 64, 2048 / 256), 256, 0, stream>>>(
      x1b, wstg, nullptr, qkvz, 2048, 3072, 1024);
  ba_kernel<<<cdiv(2048 * 16, T), T, 0, stream>>>(x1f, Wba, bab);
  conv_silu_kernel<<<cdiv(2048 * 2048, T), T, 0, stream>>>(qkvz, convw, mixed);
  prep_kernel<<<cdiv(2048 * 8, 128), 128, 0, stream>>>(mixed, bab, Alog, dtb, qn, kn, gv, bv);
  delta_kernel<<<8, 128, 0, stream>>>(qn, kn, mixed, gv, bv, coreb);
  gatenorm_kernel<<<2048 * 8, 128, 0, stream>>>(coreb, qkvz, gnw, coreB);
  pack_b_kernel<<<cdiv(1024 * 1024, T), T, 0, stream>>>(Wout, wstg, 1024, 1024);
  gemm_bf16_wmma<true><<<dim3(1024 / 64, 2048 / 256), 256, 0, stream>>>(
      coreB, wstg, hs /* +resid */, hbuf, 2048, 1024, 1024);

  // ---- MoE sub-block ----
  zrms_kernel<<<2048, 256, 0, stream>>>(hbuf, wln2, x1f, x1b);  // x2
  router_kernel<<<cdiv(2048, T), T, 0, stream>>>(x1f, Wr, route, topi);
  copy_kernel<<<cdiv(2048 * 1024, T), T, 0, stream>>>(hbuf, outf, 2048 * 1024);
  for (int e = 0; e < 8; ++e) {
    pack_b_kernel<<<cdiv(1024 * 512, T), T, 0, stream>>>(Wg + (size_t)e * 1024 * 512, wstg, 1024, 512);
    gemm_bf16_wmma<false><<<dim3(512 / 64, 2048 / 256), 256, 0, stream>>>(
        x1b, wstg, nullptr, gbuf, 2048, 512, 1024);
    pack_b_kernel<<<cdiv(1024 * 512, T), T, 0, stream>>>(Wu + (size_t)e * 1024 * 512, wstg, 1024, 512);
    gemm_bf16_wmma<false><<<dim3(512 / 64, 2048 / 256), 256, 0, stream>>>(
        x1b, wstg, nullptr, ubuf, 2048, 512, 1024);
    silu_mul_kernel<<<cdiv(2048 * 512, T), T, 0, stream>>>(gbuf, ubuf, tmpb, 2048 * 512);
    pack_b_kernel<<<cdiv(512 * 1024, T), T, 0, stream>>>(Wd + (size_t)e * 512 * 1024, wstg, 512, 1024);
    gemm_bf16_wmma<false><<<dim3(1024 / 64, 2048 / 256), 256, 0, stream>>>(
        tmpb, wstg, nullptr, expb, 2048, 1024, 512);
    accum_kernel<<<cdiv(2048 * 1024, T), T, 0, stream>>>(outf, expb, route, e);
  }
}
